// Decoder_88622355185876
// MI455X (gfx1250) — matrix operand order repaired and checked by reading, not compiled, not run on hardware
//
#include <hip/hip_runtime.h>
#include <hip/hip_bf16.h>
#include <math.h>

// ---------------- problem constants ----------------
#define Bn    4096
#define Sn    25
#define Hn    128
#define En    128
#define NFn   10
#define NGn   5
#define NFE   138     // NF + E (cur / last_node width)
#define CURP  160     // NFE padded to mult of 32
#define RSn   409     // 3H + 5 + 2NF
#define RSP   416     // RS padded to mult of 32
#define Z4H   512     // 4*H
#define CTANH 10.0f
#define NEGBIG (-1e30f)

typedef __attribute__((ext_vector_type(16))) _Float16 v16h;
typedef __attribute__((ext_vector_type(8)))  _Float16 v8h;
typedef __attribute__((ext_vector_type(4)))  _Float16 v4h;
typedef __attribute__((ext_vector_type(8)))  float    v8f;

// ---------------- fast math (avoid IEEE div / libm tanh chains) ----------------
__device__ __forceinline__ float fast_rcp(float x) { return __builtin_amdgcn_rcpf(x); }
#if __has_builtin(__builtin_amdgcn_tanhf)
__device__ __forceinline__ float fast_tanh(float x) { return __builtin_amdgcn_tanhf(x); }
#else
__device__ __forceinline__ float fast_tanh(float x) {
  float t = __expf(2.0f * x);
  return (t - 1.0f) * fast_rcp(t + 1.0f);
}
#endif
__device__ __forceinline__ float fast_sigmoid(float x) {
  return fast_rcp(1.0f + __expf(-x));
}
__device__ __forceinline__ float wave_sum(float v) {
  #pragma unroll
  for (int o = 16; o > 0; o >>= 1) v += __shfl_xor(v, o);
  return v;
}
__device__ __forceinline__ float wave_max(float v) {
  #pragma unroll
  for (int o = 16; o > 0; o >>= 1) v = fmaxf(v, __shfl_xor(v, o));
  return v;
}

// two 16B (8-half) loads -> one 16-half fragment; pure B128 VMEM, no cvt VALU
__device__ __forceinline__ v16h load16(const _Float16* __restrict__ p0,
                                       const _Float16* __restrict__ p1)
{
  v8h lo = *(const v8h*)p0;
  v8h hi = *(const v8h*)p1;
  v16h r;
  #pragma unroll
  for (int e = 0; e < 8; ++e) { r[e] = lo[e]; r[8 + e] = hi[e]; }
  return r;
}

// ---------------- WMMA GEMM (compile-time K, f16 operands, branch-free epilogue) -----
// C[M,N] = act(A1[M,K1]*W1[N,K1]^T (+ A2[M,K2]*W2[N,K2]^T) + bias)
// one 16x16 tile per wave, 4 waves/block, fully unrolled K.
// ACT: 0=none 1=relu 2=sigmoid.  HOUT: f16 output.
template<int K1, int K2, int ACT, bool HOUT>
__global__ void gemm_wmma(const _Float16* __restrict__ A1, int lda1,
                          const _Float16* __restrict__ W1, int ldw1,
                          const _Float16* __restrict__ A2, int lda2,
                          const _Float16* __restrict__ W2, int ldw2,
                          const float* __restrict__ bias,
                          void* __restrict__ Cout, int ldc, int M, int N)
{
  const int wave = threadIdx.x >> 5;
  const int lane = threadIdx.x & 31;
  const int nT = N >> 4;
  const int tiles = (M >> 4) * nT;
  const int tile = blockIdx.x * (blockDim.x >> 5) + wave;
  if (tile >= tiles) return;                 // wave-uniform: EXEC stays all-ones
  const int m0 = (tile / nT) << 4;
  const int n0 = (tile % nT) << 4;
  const int half = lane >> 4;
  const int l16 = lane & 15;
  const int mrow = m0 + l16;
  const int ncol = n0 + l16;

  v8f acc = {0.f, 0.f, 0.f, 0.f, 0.f, 0.f, 0.f, 0.f};

  const _Float16* a1row = A1 + (size_t)mrow * lda1;
  const _Float16* w1row = W1 + (size_t)ncol * ldw1;
  #pragma unroll
  for (int k = 0; k < K1; k += 32) {
    // A 16x32 frag: lane holds row mrow, K = k + half*8 + {0..7} and +16
    v16h a = load16(a1row + k + half * 8, a1row + k + half * 8 + 16);
    // B 32x16 frag: lane holds col ncol, K = k + half*8 + {0..7} and +16; B[k][n] = W[n][k]
    v16h b = load16(w1row + k + half * 8, w1row + k + half * 8 + 16);
    acc = __builtin_amdgcn_wmma_f32_16x16x32_f16(false, a, false, b, (short)0, acc,
                                                 false, false);
  }
  if constexpr (K2 > 0) {
    const _Float16* a2row = A2 + (size_t)mrow * lda2;
    const _Float16* w2row = W2 + (size_t)ncol * ldw2;
    #pragma unroll
    for (int k = 0; k < K2; k += 32) {
      v16h a = load16(a2row + k + half * 8, a2row + k + half * 8 + 16);
      v16h b = load16(w2row + k + half * 8, w2row + k + half * 8 + 16);
      acc = __builtin_amdgcn_wmma_f32_16x16x32_f16(false, a, false, b, (short)0, acc,
                                                   false, false);
    }
  }

  const float bs = bias[ncol];
  #pragma unroll
  for (int r = 0; r < 8; ++r) {
    const int m = m0 + r + 8 * half;
    float v = acc[r] + bs;
    if constexpr (ACT == 1) v = fmaxf(v, 0.0f);
    if constexpr (ACT == 2) v = fast_sigmoid(v);
    if constexpr (HOUT) ((_Float16*)Cout)[(size_t)m * ldc + ncol] = (_Float16)v;
    else                ((float*)Cout)[(size_t)m * ldc + ncol] = v;
  }
}

template<int K1, int K2, int ACT, bool HOUT>
static inline void launch_gemm(const _Float16* A1, int lda1, const _Float16* W1, int ldw1,
                               const _Float16* A2, int lda2, const _Float16* W2, int ldw2,
                               const float* bias, void* Cout, int ldc, int M, int N,
                               hipStream_t stream)
{
  int tiles = (M / 16) * (N / 16);
  gemm_wmma<K1, K2, ACT, HOUT><<<(tiles + 3) / 4, 128, 0, stream>>>(
      A1, lda1, W1, ldw1, A2, lda2, W2, ldw2, bias, Cout, ldc, M, N);
}

// ---------------- f32 -> f16 convert (flat) ----------------
__global__ void cvt_f16(const float* __restrict__ src, _Float16* __restrict__ dst, int n)
{
  int t = blockIdx.x * blockDim.x + threadIdx.x;
  if (t < n) dst[t] = (_Float16)src[t];
}

// ---------------- f32 [rows,cols] -> zero-padded f16 [rows,ldd] ----------------
__global__ void pad_cvt(const float* __restrict__ src, _Float16* __restrict__ dst,
                        int rows, int cols, int ldd)
{
  int t = blockIdx.x * blockDim.x + threadIdx.x;
  if (t >= rows * ldd) return;
  int r = t / ldd, c = t % ldd;
  dst[t] = (c < cols) ? (_Float16)src[r * cols + c] : (_Float16)0.0f;
}

// ---------------- per-call state init ----------------
__global__ void init_state(const float* __restrict__ h0, const float* __restrict__ c0,
                           const float* __restrict__ dec, const float* __restrict__ fni,
                           _Float16* __restrict__ h, float* __restrict__ c,
                           _Float16* __restrict__ he, float* __restrict__ ce,
                           _Float16* __restrict__ x, _Float16* __restrict__ curp,
                           _Float16* __restrict__ last)
{
  int t = blockIdx.x * blockDim.x + threadIdx.x;
  if (t < Bn * Hn) {
    h[t] = (_Float16)h0[t]; he[t] = (_Float16)h0[t];
    c[t] = c0[t];           ce[t] = c0[t];
    x[t] = (_Float16)dec[t];
  }
  if (t < Bn * CURP) {
    int b = t / CURP, j = t % CURP;
    curp[t] = (j < NFE) ? (_Float16)fni[(size_t)b * NFE + j] : (_Float16)0.0f;
  }
  if (t < Bn * NFE) last[t] = (_Float16)fni[t];
}

// ---------------- LSTM gates; h -> f16 (GEMM operand), c stays f32 ------------------
__global__ void lstm_gates(const float* __restrict__ z, _Float16* __restrict__ h,
                           float* __restrict__ c)
{
  int t = blockIdx.x * blockDim.x + threadIdx.x;
  if (t >= Bn * Hn) return;
  int b = t / Hn, j = t % Hn;
  const float* zr = z + (size_t)b * Z4H;
  float gi = fast_sigmoid(zr[j]);
  float gf = fast_sigmoid(zr[j + Hn]);
  float gg = fast_tanh(zr[j + 2 * Hn]);
  float go = fast_sigmoid(zr[j + 3 * Hn]);
  float c2 = gf * c[t] + gi * gg;
  c[t] = c2;
  h[t] = (_Float16)(go * fast_tanh(c2));
}

// ---------------- mask: init/update + mask_modify -----------------------------------
__global__ void mask_update(const unsigned char* __restrict__ reach,
                            const int* __restrict__ prev, int* __restrict__ mask, int step)
{
  int b = blockIdx.x * blockDim.x + threadIdx.x;
  if (b >= Bn) return;
  int* m = mask + (size_t)b * Sn;
  if (step == 0) {
    #pragma unroll
    for (int s = 0; s < Sn; ++s) m[s] = reach[(size_t)b * Sn + s] ? 1 : 0;
  } else {
    m[prev[b]] = 1;
  }
  int all = 1;
  #pragma unroll
  for (int s = 0; s < Sn; ++s) all &= m[s];
  if (all) m[Sn - 1] = 0;
}

// ---------------- u[b,s] = sum_h v[h]*tanh(q[b,h] + e[s,b,h]) ; one wave per (s,b) ---
__global__ void attn_score(const _Float16* __restrict__ e, const float* __restrict__ q,
                           const float* __restrict__ vvec, float* __restrict__ u)
{
  int wave = threadIdx.x >> 5, lane = threadIdx.x & 31;
  int p = blockIdx.x * (blockDim.x >> 5) + wave;     // p = s*B + b
  if (p >= Bn * Sn) return;
  int s = p / Bn, b = p % Bn;
  v4h   ev = ((const v4h*)(e + (size_t)p * Hn))[lane];
  float4 qv = ((const float4*)(q + (size_t)b * Hn))[lane];
  float4 vv = ((const float4*)vvec)[lane];
  float acc = vv.x * fast_tanh(qv.x + (float)ev[0])
            + vv.y * fast_tanh(qv.y + (float)ev[1])
            + vv.z * fast_tanh(qv.z + (float)ev[2])
            + vv.w * fast_tanh(qv.w + (float)ev[3]);
  acc = wave_sum(acc);
  if (lane == 0) u[(size_t)b * Sn + s] = acc;
}

// ---------------- masked softmax over S (glimpse weights); one wave per b -----------
__global__ void glp_softmax(const float* __restrict__ u, const int* __restrict__ mask,
                            float* __restrict__ p)
{
  int wave = threadIdx.x >> 5, lane = threadIdx.x & 31;
  int b = blockIdx.x * (blockDim.x >> 5) + wave;
  if (b >= Bn) return;
  float v = NEGBIG;
  if (lane < Sn && !mask[(size_t)b * Sn + lane]) v = u[(size_t)b * Sn + lane];
  float m = wave_max(v);
  float ex = (v > NEGBIG * 0.5f) ? __expf(v - m) : 0.0f;
  float sum = wave_sum(ex);
  float rs = fast_rcp(sum);
  if (lane < Sn) p[(size_t)b * Sn + lane] = ex * rs;
}

// ---------------- g[b,h] = sum_s e[s,b,h] * p[b,s]  (f16 out: GEMM operand) ---------
__global__ void glimpse_comb(const _Float16* __restrict__ e, const float* __restrict__ p,
                             _Float16* __restrict__ g)
{
  int t = blockIdx.x * blockDim.x + threadIdx.x;
  if (t >= Bn * Hn) return;
  int b = t / Hn, hh = t % Hn;
  float acc = 0.0f;
  #pragma unroll
  for (int s = 0; s < Sn; ++s)
    acc += (float)e[((size_t)s * Bn + b) * Hn + hh] * p[(size_t)b * Sn + s];
  g[t] = (_Float16)acc;
}

// ---------------- pointer head: 10*tanh(u), mask, log_softmax, greedy argmax --------
__global__ void ptr_select(const float* __restrict__ u, const int* __restrict__ mask,
                           float* __restrict__ logp_out, float* __restrict__ sel_out,
                           int* __restrict__ idx_out)
{
  int wave = threadIdx.x >> 5, lane = threadIdx.x & 31;
  int b = blockIdx.x * (blockDim.x >> 5) + wave;
  if (b >= Bn) return;
  float lv = NEGBIG;
  if (lane < Sn && !mask[(size_t)b * Sn + lane])
    lv = CTANH * fast_tanh(u[(size_t)b * Sn + lane]);
  float m = wave_max(lv);
  float ex = (lv > NEGBIG * 0.5f) ? __expf(lv - m) : 0.0f;
  float sum = wave_sum(ex);
  float lse = m + __logf(sum);
  if (lane < Sn) logp_out[(size_t)b * Sn + lane] = lv - lse;
  // argmax (first maximum)
  float bv = lv; int bi = lane;
  #pragma unroll
  for (int o = 16; o > 0; o >>= 1) {
    float ov = __shfl_xor(bv, o);
    int   oi = __shfl_xor(bi, o);
    if (ov > bv || (ov == bv && oi < bi)) { bv = ov; bi = oi; }
  }
  if (lane == 0) { idx_out[b] = bi; sel_out[b] = (float)bi; }
}

// ---------------- gather cur = [V_mask[b,idx], context[idx,b]] + next decoder input --
__global__ void gather_step(const int* __restrict__ idx, const float* __restrict__ Vmask,
                            const _Float16* __restrict__ ctx16,
                            const float* __restrict__ embedded,
                            _Float16* __restrict__ curp, _Float16* __restrict__ x)
{
  int b = blockIdx.x;
  int t = threadIdx.x;            // blockDim = 192
  int id = idx[b];
  if (t < NFn)       curp[(size_t)b * CURP + t] = (_Float16)Vmask[((size_t)b * Sn + id) * NFn + t];
  else if (t < NFE)  curp[(size_t)b * CURP + t] = ctx16[((size_t)id * Bn + b) * Hn + (t - NFn)];
  else if (t < CURP) curp[(size_t)b * CURP + t] = (_Float16)0.0f;
  if (t < En) x[(size_t)b * En + t] = (_Float16)embedded[((size_t)id * Bn + b) * En + t];
}

// ---------------- route_state = [start(5) | last(138) | cur(138) | he(128) | pad] ----
__global__ void route_build(const float* __restrict__ start, const _Float16* __restrict__ curp,
                            const _Float16* __restrict__ he, _Float16* __restrict__ last,
                            _Float16* __restrict__ rs)
{
  int b = blockIdx.x;
  int t = threadIdx.x;            // blockDim = 128
  for (int j = t; j < RSP; j += blockDim.x) {
    _Float16 v;
    if (j < 5)                 v = (_Float16)start[(size_t)b * 5 + j];
    else if (j < 5 + NFE)      v = last[(size_t)b * NFE + (j - 5)];
    else if (j < 5 + 2 * NFE)  v = curp[(size_t)b * CURP + (j - 5 - NFE)];
    else if (j < RSn)          v = he[(size_t)b * Hn + (j - 5 - 2 * NFE)];
    else                       v = (_Float16)0.0f;
    rs[(size_t)b * RSP + j] = v;
  }
  __syncthreads();
  for (int j = t; j < NFE; j += blockDim.x)     // last_node <- cur for next step
    last[(size_t)b * NFE + j] = curp[(size_t)b * CURP + j];
}

// ---------------- MDN output: out[b] = sum_g mu[g]*softmax(pi)[g] ; one wave per b ---
__global__ void mdn_out(const float* __restrict__ muh, const float* __restrict__ pih,
                        const float* __restrict__ Wmu2, const float* __restrict__ bmu2,
                        const float* __restrict__ Wpi2, const float* __restrict__ bpi2,
                        float* __restrict__ out)
{
  int b = blockIdx.x;
  int lane = threadIdx.x;         // blockDim = 32
  float mu[NGn], pi[NGn];
  #pragma unroll
  for (int g = 0; g < NGn; ++g) { mu[g] = 0.0f; pi[g] = 0.0f; }
  #pragma unroll
  for (int hh0 = 0; hh0 < Hn; hh0 += 32) {
    int hh = hh0 + lane;
    float mh = muh[(size_t)b * Hn + hh];
    float ph = pih[(size_t)b * Hn + hh];
    #pragma unroll
    for (int g = 0; g < NGn; ++g) {
      mu[g] += mh * Wmu2[g * Hn + hh];
      pi[g] += ph * Wpi2[g * Hn + hh];
    }
  }
  #pragma unroll
  for (int g = 0; g < NGn; ++g) { mu[g] = wave_sum(mu[g]); pi[g] = wave_sum(pi[g]); }
  if (lane == 0) {
    float mx = NEGBIG;
    #pragma unroll
    for (int g = 0; g < NGn; ++g) {
      mu[g] += bmu2[g];
      pi[g] += bpi2[g];
      mx = fmaxf(mx, pi[g]);
    }
    float s = 0.0f, acc = 0.0f;
    #pragma unroll
    for (int g = 0; g < NGn; ++g) {
      float e = __expf(pi[g] - mx);
      s += e;
      acc += mu[g] * e;
    }
    out[b] = acc * fast_rcp(s);
  }
}

// =====================================================================================
extern "C" void kernel_launch(void* const* d_in, const int* in_sizes, int n_in,
                              void* d_out, int out_size, void* d_ws, size_t ws_size,
                              hipStream_t stream)
{
  (void)in_sizes; (void)n_in; (void)out_size; (void)ws_size;

  // -------- inputs (setup_inputs dict order; params dict flattened in order) --------
  const float* start_fea  = (const float*)d_in[0];
  const float* dec_in     = (const float*)d_in[1];
  const float* embedded   = (const float*)d_in[2];
  const float* h0         = (const float*)d_in[3];
  const float* c0         = (const float*)d_in[4];
  const float* context    = (const float*)d_in[5];
  const unsigned char* reach = (const unsigned char*)d_in[6];
  const float* Vmask      = (const float*)d_in[7];
  const float* fni        = (const float*)d_in[8];
  const float* lstm_Wih   = (const float*)d_in[9];
  const float* lstm_Whh   = (const float*)d_in[10];
  const float* lstm_b     = (const float*)d_in[11];
  const float* eta_Wih    = (const float*)d_in[12];
  const float* eta_Whh    = (const float*)d_in[13];
  const float* eta_b      = (const float*)d_in[14];
  const float* etamlp_W   = (const float*)d_in[15];
  const float* etamlp_b   = (const float*)d_in[16];
  const float* sigmlp_W   = (const float*)d_in[17];
  const float* sigmlp_b   = (const float*)d_in[18];
  const float* glp_Wq = (const float*)d_in[19]; const float* glp_bq = (const float*)d_in[20];
  const float* glp_Wr = (const float*)d_in[21]; const float* glp_br = (const float*)d_in[22];
  const float* glp_v  = (const float*)d_in[23];
  const float* ptr_Wq = (const float*)d_in[24]; const float* ptr_bq = (const float*)d_in[25];
  const float* ptr_Wr = (const float*)d_in[26]; const float* ptr_br = (const float*)d_in[27];
  const float* ptr_v  = (const float*)d_in[28];
  const float* emu_W1 = (const float*)d_in[29]; const float* emu_b1 = (const float*)d_in[30];
  const float* emu_W2 = (const float*)d_in[31]; const float* emu_b2 = (const float*)d_in[32];
  const float* epi_W1 = (const float*)d_in[33]; const float* epi_b1 = (const float*)d_in[34];
  const float* epi_W2 = (const float*)d_in[35]; const float* epi_b2 = (const float*)d_in[36];
  const float* smu_W1 = (const float*)d_in[37]; const float* smu_b1 = (const float*)d_in[38];
  const float* smu_W2 = (const float*)d_in[39]; const float* smu_b2 = (const float*)d_in[40];
  const float* spi_W1 = (const float*)d_in[41]; const float* spi_b1 = (const float*)d_in[42];
  const float* spi_W2 = (const float*)d_in[43]; const float* spi_b2 = (const float*)d_in[44];

  // -------- outputs: log_ps [S,B,S] | sels [S,B] | etas [S,B] | sigmas [S,B] --------
  float* out      = (float*)d_out;
  float* out_logp = out;
  float* out_sel  = out + (size_t)Sn * Bn * Sn;
  float* out_eta  = out_sel + (size_t)Sn * Bn;
  float* out_sig  = out_eta + (size_t)Sn * Bn;

  // -------- workspace layout --------
  size_t off = 0;
  char* wsb = (char*)d_ws;
  auto alloc = [&](size_t nbytes) -> void* {
    off = (off + 255) & ~(size_t)255;
    void* p = wsb + off;
    off += nbytes;
    return p;
  };
  const size_t SBH = (size_t)Sn * Bn * Hn;
  _Float16* eglp  = (_Float16*)alloc(SBH * 2);
  _Float16* eptr  = (_Float16*)alloc(SBH * 2);
  _Float16* ctx16 = (_Float16*)alloc(SBH * 2);
  float* z    = (float*)alloc((size_t)Bn * Z4H * 4);
  float* ze   = (float*)alloc((size_t)Bn * Z4H * 4);
  _Float16* h    = (_Float16*)alloc((size_t)Bn * Hn * 2);
  _Float16* he   = (_Float16*)alloc((size_t)Bn * Hn * 2);
  _Float16* x    = (_Float16*)alloc((size_t)Bn * Hn * 2);
  _Float16* g    = (_Float16*)alloc((size_t)Bn * Hn * 2);
  _Float16* etah = (_Float16*)alloc((size_t)Bn * Hn * 2);
  _Float16* sigh = (_Float16*)alloc((size_t)Bn * Hn * 2);
  _Float16* curp = (_Float16*)alloc((size_t)Bn * CURP * 2);
  _Float16* rs   = (_Float16*)alloc((size_t)Bn * RSP * 2);
  _Float16* last = (_Float16*)alloc((size_t)Bn * NFE * 2);
  float* c    = (float*)alloc((size_t)Bn * Hn * 4);
  float* ce   = (float*)alloc((size_t)Bn * Hn * 4);
  float* q    = (float*)alloc((size_t)Bn * Hn * 4);
  float* q2   = (float*)alloc((size_t)Bn * Hn * 4);
  float* muh  = (float*)alloc((size_t)Bn * Hn * 4);
  float* pih  = (float*)alloc((size_t)Bn * Hn * 4);
  float* u    = (float*)alloc((size_t)Bn * Sn * 4);
  float* u2   = (float*)alloc((size_t)Bn * Sn * 4);
  float* pr   = (float*)alloc((size_t)Bn * Sn * 4);
  int*   mask = (int*)alloc((size_t)Bn * Sn * 4);
  int*   idx  = (int*)alloc((size_t)Bn * 4);
  // f16 (zero-padded where needed) weights
  _Float16* wWih    = (_Float16*)alloc((size_t)Z4H * Hn * 2);
  _Float16* wWhh    = (_Float16*)alloc((size_t)Z4H * Hn * 2);
  _Float16* wEtaWih = (_Float16*)alloc((size_t)Z4H * CURP * 2);
  _Float16* wEtaWhh = (_Float16*)alloc((size_t)Z4H * Hn * 2);
  _Float16* wGlpWq  = (_Float16*)alloc((size_t)Hn * Hn * 2);
  _Float16* wPtrWq  = (_Float16*)alloc((size_t)Hn * Hn * 2);
  _Float16* wGlpWr  = (_Float16*)alloc((size_t)Hn * Hn * 2);
  _Float16* wPtrWr  = (_Float16*)alloc((size_t)Hn * Hn * 2);
  _Float16* wEtaMlp = (_Float16*)alloc((size_t)Hn * RSP * 2);
  _Float16* wSigMlp = (_Float16*)alloc((size_t)Hn * RSP * 2);
  _Float16* wEmu1   = (_Float16*)alloc((size_t)Hn * Hn * 2);
  _Float16* wEpi1   = (_Float16*)alloc((size_t)Hn * Hn * 2);
  _Float16* wSmu1   = (_Float16*)alloc((size_t)Hn * Hn * 2);
  _Float16* wSpi1   = (_Float16*)alloc((size_t)Hn * Hn * 2);

  // -------- one-time per call: weight cvt/pad, context cvt, state init ---------------
  auto padw = [&](const float* src, _Float16* dst, int rows, int cols, int ldd) {
    pad_cvt<<<((rows * ldd) + 255) / 256, 256, 0, stream>>>(src, dst, rows, cols, ldd);
  };
  padw(lstm_Wih, wWih,    Z4H, Hn,  Hn);
  padw(lstm_Whh, wWhh,    Z4H, Hn,  Hn);
  padw(eta_Wih,  wEtaWih, Z4H, NFE, CURP);
  padw(eta_Whh,  wEtaWhh, Z4H, Hn,  Hn);
  padw(glp_Wq,   wGlpWq,  Hn,  Hn,  Hn);
  padw(ptr_Wq,   wPtrWq,  Hn,  Hn,  Hn);
  padw(glp_Wr,   wGlpWr,  Hn,  Hn,  Hn);
  padw(ptr_Wr,   wPtrWr,  Hn,  Hn,  Hn);
  padw(etamlp_W, wEtaMlp, Hn,  RSn, RSP);
  padw(sigmlp_W, wSigMlp, Hn,  RSn, RSP);
  padw(emu_W1,   wEmu1,   Hn,  Hn,  Hn);
  padw(epi_W1,   wEpi1,   Hn,  Hn,  Hn);
  padw(smu_W1,   wSmu1,   Hn,  Hn,  Hn);
  padw(spi_W1,   wSpi1,   Hn,  Hn,  Hn);
  cvt_f16<<<((int)SBH + 255) / 256, 256, 0, stream>>>(context, ctx16, (int)SBH);
  init_state<<<((Bn * CURP) + 255) / 256, 256, 0, stream>>>(h0, c0, dec_in, fni,
                                                            h, c, he, ce, x, curp, last);
  // e[s*B+b, o] = context[s,b,:] . Wr[o,:] + br   (step-invariant; f16 in/out)
  launch_gemm<Hn, 0, 0, true>(ctx16, Hn, wGlpWr, Hn, nullptr, 0, nullptr, 0,
                              glp_br, eglp, Hn, Sn * Bn, Hn, stream);
  launch_gemm<Hn, 0, 0, true>(ctx16, Hn, wPtrWr, Hn, nullptr, 0, nullptr, 0,
                              ptr_br, eptr, Hn, Sn * Bn, Hn, stream);
  // he0, ce0 = lstm_eta(first_node_input, h0, c0)   (fused dual-A GEMM)
  launch_gemm<CURP, Hn, 0, false>(curp, CURP, wEtaWih, CURP, he, Hn, wEtaWhh, Hn,
                                  eta_b, ze, Z4H, Bn, Z4H, stream);
  lstm_gates<<<((Bn * Hn) + 255) / 256, 256, 0, stream>>>(ze, he, ce);

  // -------- decode loop --------
  for (int i = 0; i < Sn; ++i) {
    mask_update<<<(Bn + 255) / 256, 256, 0, stream>>>(reach, idx, mask, i);
    // pointer LSTM: z = x*Wih^T + h*Whh^T + b (fused)
    launch_gemm<Hn, Hn, 0, false>(x, Hn, wWih, Hn, h, Hn, wWhh, Hn,
                                  lstm_b, z, Z4H, Bn, Z4H, stream);
    lstm_gates<<<((Bn * Hn) + 255) / 256, 256, 0, stream>>>(z, h, c);
    // glimpse attention
    launch_gemm<Hn, 0, 0, false>(h, Hn, wGlpWq, Hn, nullptr, 0, nullptr, 0,
                                 glp_bq, q, Hn, Bn, Hn, stream);
    attn_score<<<((Bn * Sn) + 3) / 4, 128, 0, stream>>>(eglp, q, glp_v, u);
    glp_softmax<<<(Bn + 3) / 4, 128, 0, stream>>>(u, mask, pr);
    glimpse_comb<<<((Bn * Hn) + 255) / 256, 256, 0, stream>>>(eglp, pr, g);
    // pointer attention + greedy select
    launch_gemm<Hn, 0, 0, false>(g, Hn, wPtrWq, Hn, nullptr, 0, nullptr, 0,
                                 ptr_bq, q2, Hn, Bn, Hn, stream);
    attn_score<<<((Bn * Sn) + 3) / 4, 128, 0, stream>>>(eptr, q2, ptr_v, u2);
    ptr_select<<<(Bn + 3) / 4, 128, 0, stream>>>(u2, mask,
                                                 out_logp + (size_t)i * Bn * Sn,
                                                 out_sel + (size_t)i * Bn, idx);
    // gather selected node feats / next decoder input
    gather_step<<<Bn, 192, 0, stream>>>(idx, Vmask, ctx16, embedded, curp, x);
    // eta LSTM (fused dual-A)
    launch_gemm<CURP, Hn, 0, false>(curp, CURP, wEtaWih, CURP, he, Hn, wEtaWhh, Hn,
                                    eta_b, ze, Z4H, Bn, Z4H, stream);
    lstm_gates<<<((Bn * Hn) + 255) / 256, 256, 0, stream>>>(ze, he, ce);
    // route state + MDN heads (eta)
    route_build<<<Bn, 128, 0, stream>>>(start_fea, curp, he, last, rs);
    launch_gemm<RSP, 0, 1, true>(rs, RSP, wEtaMlp, RSP, nullptr, 0, nullptr, 0,
                                 etamlp_b, etah, Hn, Bn, Hn, stream);
    launch_gemm<Hn, 0, 2, false>(etah, Hn, wEmu1, Hn, nullptr, 0, nullptr, 0,
                                 emu_b1, muh, Hn, Bn, Hn, stream);
    launch_gemm<Hn, 0, 1, false>(etah, Hn, wEpi1, Hn, nullptr, 0, nullptr, 0,
                                 epi_b1, pih, Hn, Bn, Hn, stream);
    mdn_out<<<Bn, 32, 0, stream>>>(muh, pih, emu_W2, emu_b2, epi_W2, epi_b2,
                                   out_eta + (size_t)i * Bn);
    // MDN heads (sigma)
    launch_gemm<RSP, 0, 1, true>(rs, RSP, wSigMlp, RSP, nullptr, 0, nullptr, 0,
                                 sigmlp_b, sigh, Hn, Bn, Hn, stream);
    launch_gemm<Hn, 0, 2, false>(sigh, Hn, wSmu1, Hn, nullptr, 0, nullptr, 0,
                                 smu_b1, muh, Hn, Bn, Hn, stream);
    launch_gemm<Hn, 0, 1, false>(sigh, Hn, wSpi1, Hn, nullptr, 0, nullptr, 0,
                                 spi_b1, pih, Hn, Bn, Hn, stream);
    mdn_out<<<Bn, 32, 0, stream>>>(muh, pih, smu_W2, smu_b2, spi_W2, spi_b2,
                                   out_sig + (size_t)i * Bn);
  }
}
